// DynamicConvolution_3264175145732
// MI455X (gfx1250) — compile-verified, matrix operand order checked by reading, not run on hardware
//
#include <hip/hip_runtime.h>

// ---------------- problem constants ----------------
#define BB   8
#define TT   2048
#define CC   512
#define HH   8
#define KK   31
#define PADW 15
#define MM   (BB * TT)   // 16384 rows
#define HK   (HH * KK)   // 248
#define NPAD 256         // padded H*K for clean WMMA tiling

// ---------------- vector types ----------------
typedef __attribute__((ext_vector_type(16))) __bf16 v16bf;
typedef __attribute__((ext_vector_type(8)))  __bf16 v8bf;
typedef __attribute__((ext_vector_type(8)))  float  v8f;
typedef int v4i_gcc __attribute__((vector_size(16)));   // matches builtin param

// Load a bf16 WMMA fragment (A 16x32 row-major, or B as rows of W) for one lane.
// lanes 0-15: K=[k0..k0+7] and [k0+16..k0+23]; lanes 16-31: +8.  Two 16B loads.
// Works for both global and LDS pointers (compiler picks global_load / ds_load).
__device__ __forceinline__ v16bf load_frag(const __bf16* rowptr, int k0, int hi) {
  union { v16bf v; v8bf h[2]; } u;
  u.h[0] = *(const v8bf*)(rowptr + k0 + hi * 8);
  u.h[1] = *(const v8bf*)(rowptr + k0 + 16 + hi * 8);
  return u.v;
}

__device__ __forceinline__ v8f wmma_bf16(v16bf a, v16bf b, v8f c) {
  return __builtin_amdgcn_wmma_f32_16x16x32_bf16(false, a, false, b,
                                                 (short)0, c, false, false);
}

// Stage one contiguous 16KB A-stripe (16 rows x 512 cols bf16) into LDS.
// Uses gfx1250 async global->LDS copies (ASYNCcnt) when available.
__device__ __forceinline__ void stage_A(const __bf16* gsrc, __bf16* sdst, int tid) {
#if __has_builtin(__builtin_amdgcn_global_load_async_to_lds_b128) && \
    __has_builtin(__builtin_amdgcn_s_wait_asynccnt)
  #pragma unroll
  for (int i = 0; i < 4; ++i) {
    int off = (tid + i * 256) * 8;   // 8 bf16 = 16 bytes per lane per issue
    __builtin_amdgcn_global_load_async_to_lds_b128(
        (__attribute__((address_space(1))) v4i_gcc*)(gsrc + off),
        (__attribute__((address_space(3))) v4i_gcc*)(sdst + off), 0, 0);
  }
  __builtin_amdgcn_s_wait_asynccnt(0);
#else
  #pragma unroll
  for (int i = 0; i < 4; ++i) {
    int off = (tid + i * 256) * 8;
    *(v8bf*)(sdst + off) = *(const v8bf*)(gsrc + off);
  }
#endif
  __syncthreads();
}

// ---------------- f32 -> bf16 conversion ----------------
__global__ void cvt_f32_bf16(const float* __restrict__ src,
                             __bf16* __restrict__ dst, int n) {
  int i = blockIdx.x * blockDim.x + threadIdx.x;
  int stride = gridDim.x * blockDim.x;
  for (; i < n; i += stride) dst[i] = (__bf16)src[i];
}

// wlw (248x512) -> zero-padded (256x512) bf16
__global__ void cvt_pad_wlw(const float* __restrict__ src,
                            __bf16* __restrict__ dst) {
  int i = blockIdx.x * blockDim.x + threadIdx.x;
  if (i >= NPAD * CC) return;
  int row = i >> 9;  // /512
  dst[i] = (row < HK) ? (__bf16)src[i] : (__bf16)0.0f;
}

// ---------------- GEMM1 + GLU:  x = a * sigmoid(g) ----------------
// One block = 256 threads (8 waves) owns a 16-row stripe; A staged in LDS once.
// Each wave owns 64 output cols (4 tiles) x both GLU halves -> 8 accumulators.
// All 8 B-fragments of a K-step are loaded into distinct registers BEFORE the
// WMMAs so the backend can clause the loads and use partial loadcnt waits.
__global__ void __launch_bounds__(256)
gemm_glu_kernel(const __bf16* __restrict__ qb, const __bf16* __restrict__ w1b,
                const float* __restrict__ b1, __bf16* __restrict__ xb) {
  __shared__ __align__(16) __bf16 sA[16 * CC];  // 16 KB
  const int tid  = threadIdx.x;
  const int wave = tid >> 5;
  const int lane = tid & 31;
  const int m_lo = lane & 15;
  const int hi   = lane >> 4;
  const int row0 = blockIdx.x * 16;

  stage_A(qb + row0 * CC, sA, tid);

  const __bf16* srow = sA + m_lo * CC;
  const int colbase = wave * 64;

  const __bf16* browA[4];
  const __bf16* browG[4];
  #pragma unroll
  for (int ct = 0; ct < 4; ++ct) {
    int c0 = colbase + ct * 16;
    browA[ct] = w1b + (c0 + m_lo) * CC;
    browG[ct] = w1b + (c0 + CC + m_lo) * CC;
  }

  v8f accA[4] = {{}, {}, {}, {}};
  v8f accG[4] = {{}, {}, {}, {}};
  for (int k0 = 0; k0 < CC; k0 += 32) {
    v16bf a = load_frag(srow, k0, hi);           // ds_load_b128 x2
    v16bf bfr[8];
    #pragma unroll
    for (int ct = 0; ct < 4; ++ct) {
      bfr[2 * ct]     = load_frag(browA[ct], k0, hi);   // L2-resident weights
      bfr[2 * ct + 1] = load_frag(browG[ct], k0, hi);
    }
    #pragma unroll
    for (int ct = 0; ct < 4; ++ct) {
      accA[ct] = wmma_bf16(a, bfr[2 * ct],     accA[ct]);
      accG[ct] = wmma_bf16(a, bfr[2 * ct + 1], accG[ct]);
    }
  }

  #pragma unroll
  for (int ct = 0; ct < 4; ++ct) {
    int col = colbase + ct * 16 + m_lo;
    float biasA = b1[col];
    float biasG = b1[col + CC];
    #pragma unroll
    for (int r = 0; r < 8; ++r) {
      int row = row0 + r + 8 * hi;
      float av = accA[ct][r] + biasA;
      float gv = accG[ct][r] + biasG;
      xb[row * CC + col] = (__bf16)(av / (1.0f + __expf(-gv)));
    }
  }
}

// ---------------- GEMM2: dynamic-kernel logits (padded N=256) ----------------
// 8 waves x 2 col-tiles = 256 cols per 16-row stripe.
__global__ void __launch_bounds__(256)
gemm_w_kernel(const __bf16* __restrict__ xb, const __bf16* __restrict__ wlwb,
              const float* __restrict__ wlb, float* __restrict__ logits) {
  __shared__ __align__(16) __bf16 sA[16 * CC];
  const int tid  = threadIdx.x;
  const int wave = tid >> 5;
  const int lane = tid & 31;
  const int m_lo = lane & 15;
  const int hi   = lane >> 4;
  const int row0 = blockIdx.x * 16;

  stage_A(xb + row0 * CC, sA, tid);

  const __bf16* srow = sA + m_lo * CC;
  const int colbase = wave * 32;

  const __bf16* brow[2];
  #pragma unroll
  for (int ct = 0; ct < 2; ++ct)
    brow[ct] = wlwb + (colbase + ct * 16 + m_lo) * CC;

  v8f acc[2] = {{}, {}};
  for (int k0 = 0; k0 < CC; k0 += 32) {
    v16bf a = load_frag(srow, k0, hi);
    v16bf bfr[2];
    #pragma unroll
    for (int ct = 0; ct < 2; ++ct) bfr[ct] = load_frag(brow[ct], k0, hi);
    #pragma unroll
    for (int ct = 0; ct < 2; ++ct) acc[ct] = wmma_bf16(a, bfr[ct], acc[ct]);
  }

  #pragma unroll
  for (int ct = 0; ct < 2; ++ct) {
    int col = colbase + ct * 16 + m_lo;
    if (col < HK) {
      float bias = wlb[col];
      #pragma unroll
      for (int r = 0; r < 8; ++r) {
        int row = row0 + r + 8 * hi;
        logits[row * NPAD + col] = acc[ct][r] + bias;
      }
    }
  }
}

// ---------------- windowed softmax over K=31 with band validity ----------------
__global__ void softmax_kernel(float* __restrict__ logits) {
  int idx = blockIdx.x * blockDim.x + threadIdx.x;  // (b*T+t)*H + h
  if (idx >= MM * HH) return;
  int row = idx >> 3;        // b*T + t
  int h   = idx & 7;
  int t   = row & (TT - 1);
  float* p = logits + row * NPAD + h * KK;

  float l[KK];
  float m = -1e30f;
  #pragma unroll
  for (int j = 0; j < KK; ++j) {
    int s = t + j - PADW;
    bool valid = (s >= 0) && (s < TT);
    float v = valid ? p[j] : -1e30f;
    l[j] = v;
    m = fmaxf(m, v);
  }
  float sum = 0.0f;
  #pragma unroll
  for (int j = 0; j < KK; ++j) {
    float e = __expf(l[j] - m);   // invalid taps underflow to 0
    l[j] = e;
    sum += e;
  }
  float inv = 1.0f / sum;
  #pragma unroll
  for (int j = 0; j < KK; ++j) p[j] = l[j] * inv;
}

// ---------------- banded conv: y[t,c] = sum_j p[t,h,j] * x[t+j-pad,c] ----------
__global__ void __launch_bounds__(256)
conv_kernel(const float* __restrict__ probs, const __bf16* __restrict__ xb,
            const int* __restrict__ mask, __bf16* __restrict__ yb) {
  __shared__ float ps[HK];
  const int row = blockIdx.x;        // b*T + t
  const int t   = row & (TT - 1);
  const int b   = row >> 11;
  const int tid = threadIdx.x;

  if (tid < HK) ps[tid] = probs[row * NPAD + tid];
  __syncthreads();

  const int mv = mask[row];
  #pragma unroll
  for (int cc = 0; cc < 2; ++cc) {
    int c = tid + cc * 256;
    int h = c >> 6;                  // c / D, D=64
    float acc = 0.0f;
    #pragma unroll
    for (int j = 0; j < KK; ++j) {
      int s = t + j - PADW;
      if (s >= 0 && s < TT) {
        acc += ps[h * KK + j] * (float)xb[(b * TT + s) * CC + c];
      }
    }
    yb[row * CC + c] = (__bf16)(mv ? acc : 0.0f);
  }
}

// ---------------- GEMM3: out = y @ w2^T + b2  (f32 out) ----------------
// 8 waves x 4 col-tiles = 512 cols per 16-row stripe.
__global__ void __launch_bounds__(256)
gemm_out_kernel(const __bf16* __restrict__ yb, const __bf16* __restrict__ w2b,
                const float* __restrict__ b2, float* __restrict__ out) {
  __shared__ __align__(16) __bf16 sA[16 * CC];
  const int tid  = threadIdx.x;
  const int wave = tid >> 5;
  const int lane = tid & 31;
  const int m_lo = lane & 15;
  const int hi   = lane >> 4;
  const int row0 = blockIdx.x * 16;

  stage_A(yb + row0 * CC, sA, tid);

  const __bf16* srow = sA + m_lo * CC;
  const int colbase = wave * 64;

  const __bf16* brow[4];
  #pragma unroll
  for (int ct = 0; ct < 4; ++ct)
    brow[ct] = w2b + (colbase + ct * 16 + m_lo) * CC;

  v8f acc[4] = {{}, {}, {}, {}};
  for (int k0 = 0; k0 < CC; k0 += 32) {
    __builtin_prefetch(brow[0] + k0 + 128, 0, 1);   // global_prefetch_b8
    v16bf a = load_frag(srow, k0, hi);
    v16bf bfr[4];
    #pragma unroll
    for (int ct = 0; ct < 4; ++ct) bfr[ct] = load_frag(brow[ct], k0, hi);
    #pragma unroll
    for (int ct = 0; ct < 4; ++ct) acc[ct] = wmma_bf16(a, bfr[ct], acc[ct]);
  }

  #pragma unroll
  for (int ct = 0; ct < 4; ++ct) {
    int col = colbase + ct * 16 + m_lo;
    float bias = b2[col];
    #pragma unroll
    for (int r = 0; r < 8; ++r) {
      int row = row0 + r + 8 * hi;
      out[row * CC + col] = acc[ct][r] + bias;
    }
  }
}

// ---------------- host launcher ----------------
extern "C" void kernel_launch(void* const* d_in, const int* in_sizes, int n_in,
                              void* d_out, int out_size, void* d_ws, size_t ws_size,
                              hipStream_t stream) {
  const float* query = (const float*)d_in[0];
  const int*   mask  = (const int*)d_in[4];
  const float* w1    = (const float*)d_in[5];
  const float* b1    = (const float*)d_in[6];
  const float* wlw   = (const float*)d_in[7];
  const float* wlb   = (const float*)d_in[8];
  const float* w2    = (const float*)d_in[9];
  const float* b2    = (const float*)d_in[10];

  char* ws = (char*)d_ws;
  size_t off = 0;
  auto alloc = [&](size_t bytes) -> char* {
    char* p = ws + off;
    off += (bytes + 255) & ~(size_t)255;
    return p;
  };
  __bf16* qb     = (__bf16*)alloc((size_t)MM * CC * 2);        // 16 MB
  __bf16* xb     = (__bf16*)alloc((size_t)MM * CC * 2);        // 16 MB
  __bf16* yb     = (__bf16*)alloc((size_t)MM * CC * 2);        // 16 MB
  __bf16* w1b    = (__bf16*)alloc((size_t)2 * CC * CC * 2);    // 1 MB
  __bf16* wlwb   = (__bf16*)alloc((size_t)NPAD * CC * 2);      // 256 KB
  __bf16* w2b    = (__bf16*)alloc((size_t)CC * CC * 2);        // 512 KB
  float*  logits = (float*)alloc((size_t)MM * NPAD * 4);       // 16 MB

  // precision conversion (bf16 feeds all WMMA GEMMs)
  cvt_f32_bf16<<<4096, 256, 0, stream>>>(query, qb, MM * CC);
  cvt_f32_bf16<<<1024, 256, 0, stream>>>(w1, w1b, 2 * CC * CC);
  cvt_f32_bf16<<<512, 256, 0, stream>>>(w2, w2b, CC * CC);
  cvt_pad_wlw<<<(NPAD * CC + 255) / 256, 256, 0, stream>>>(wlw, wlwb);

  // GEMM1 + GLU
  gemm_glu_kernel<<<MM / 16, 256, 0, stream>>>(qb, w1b, b1, xb);
  // GEMM2: dynamic kernel logits
  gemm_w_kernel<<<MM / 16, 256, 0, stream>>>(xb, wlwb, wlb, logits);
  // windowed softmax
  softmax_kernel<<<(MM * HH + 255) / 256, 256, 0, stream>>>(logits);
  // banded convolution (+ mask)
  conv_kernel<<<MM, 256, 0, stream>>>(logits, xb, mask, yb);
  // GEMM3: output projection
  gemm_out_kernel<<<MM / 16, 256, 0, stream>>>(yb, w2b, b2, (float*)d_out);
}
